// GPT3Block_2422361555283
// MI455X (gfx1250) — matrix operand order repaired and checked by reading, not compiled, not run on hardware
//
#include <hip/hip_runtime.h>
#include <math.h>

// ---------------------------------------------------------------------------
// GPT block for MI455X (gfx1250, wave32).
//   * All GEMMs: v_wmma_f32_16x16x32_bf16, 32x32 output tile per wave
//     (2 A-frags x 2 B-frags -> 4 WMMAs), fp32 accumulate.
//   * All GEMM operands live in memory as bf16; B operands are stored
//     transposed [N,K] so every fragment load is a contiguous 16B
//     vector load matching the ISA VGPR layout (CDNA5 ISA §7.12.2):
//       A frag  = two 8xbf16 chunks at K = hf*8 and K = 16+hf*8
//       B frag  = same two 8xbf16 chunks as A  (row = output column)
//   * Weights are transpose-converted once per launch through LDS tiles.
//   * Producers (LN / softmax / split / GEGLU / GEMM epilogue) emit bf16
//     mirrors for the next GEMM, fusing all conversions.
// ---------------------------------------------------------------------------

typedef __attribute__((ext_vector_type(16))) __bf16 v16bf;
typedef __attribute__((ext_vector_type(8)))  __bf16 bf16x8;
typedef __attribute__((ext_vector_type(8)))  float  v8f;

__device__ __forceinline__ v16bf load_a_frag(const __bf16* __restrict__ arow,
                                             int kk, int hf)
{
    bf16x8 a0 = *(const bf16x8*)(arow + kk + hf * 8);        // K = base .. +7
    bf16x8 a1 = *(const bf16x8*)(arow + kk + 16 + hf * 8);   // K = base+16 .. +23
    return __builtin_shufflevector(a0, a1, 0, 1, 2, 3, 4, 5, 6, 7,
                                   8, 9, 10, 11, 12, 13, 14, 15);
}

__global__ __launch_bounds__(256) void gemm_bf16_wmma(
    const __bf16* __restrict__ A,  long long strideA, int lda,   // [M,K]
    const __bf16* __restrict__ Bt, long long strideB, int ldb,   // [N,K] (B^T)
    float* __restrict__ C, __bf16* __restrict__ Cbf,             // either may be null
    long long strideC, int ldc,
    const float* __restrict__ bias, const float* __restrict__ res,
    int M, int N, int K, int batch, float alpha)
{
    const int tid  = threadIdx.x;
    const int lane = tid & 31;
    const int hf   = lane >> 4;
    const int lr   = lane & 15;

    const int mt = M >> 5, nt = N >> 5;               // 32x32 tiles
    const long long tilesPerBatch = (long long)mt * nt;
    const long long totalTiles    = tilesPerBatch * batch;
    const long long wid = (long long)blockIdx.x * (blockDim.x >> 5) + (tid >> 5);
    if (wid >= totalTiles) return;                    // wave-uniform: EXEC stays full

    const int  bb  = (int)(wid / tilesPerBatch);
    const long long rem = wid - (long long)bb * tilesPerBatch;
    const int  mi  = (int)(rem / nt);
    const int  ni  = (int)(rem - (long long)mi * nt);
    const int  m0  = mi << 5, n0 = ni << 5;

    const __bf16* Ab = A  + (long long)bb * strideA;
    const __bf16* Bb = Bt + (long long)bb * strideB;

    const __bf16* arow0 = Ab + (long long)(m0 + lr)      * lda;
    const __bf16* arow1 = Ab + (long long)(m0 + 16 + lr) * lda;
    const __bf16* brow0 = Bb + (long long)(n0 + lr)      * ldb;
    const __bf16* brow1 = Bb + (long long)(n0 + 16 + lr) * ldb;

    v8f acc00 = {}, acc01 = {}, acc10 = {}, acc11 = {};

    for (int kk = 0; kk < K; kk += 32) {
        if (kk + 32 < K) {
            __builtin_prefetch(arow0 + kk + 32, 0, 1);   // global_prefetch_b8
            __builtin_prefetch(brow0 + kk + 32, 0, 1);
        }
        const v16bf a0 = load_a_frag(arow0, kk, hf);
        const v16bf a1 = load_a_frag(arow1, kk, hf);
        const v16bf b0 = load_a_frag(brow0, kk, hf);
        const v16bf b1 = load_a_frag(brow1, kk, hf);

        acc00 = __builtin_amdgcn_wmma_f32_16x16x32_bf16(false, a0, false, b0,
                                                        (short)0, acc00, false, false);
        acc01 = __builtin_amdgcn_wmma_f32_16x16x32_bf16(false, a0, false, b1,
                                                        (short)0, acc01, false, false);
        acc10 = __builtin_amdgcn_wmma_f32_16x16x32_bf16(false, a1, false, b0,
                                                        (short)0, acc10, false, false);
        acc11 = __builtin_amdgcn_wmma_f32_16x16x32_bf16(false, a1, false, b1,
                                                        (short)0, acc11, false, false);
    }

    float*  Cb   = C   ? C   + (long long)bb * strideC : nullptr;
    __bf16* Cbfb = Cbf ? Cbf + (long long)bb * strideC : nullptr;

    const v8f accs[2][2] = {{acc00, acc01}, {acc10, acc11}};
#pragma unroll
    for (int im = 0; im < 2; ++im) {
#pragma unroll
        for (int jn = 0; jn < 2; ++jn) {
            const v8f a = accs[im][jn];
#pragma unroll
            for (int r = 0; r < 8; ++r) {
                const int row = m0 + im * 16 + r + 8 * hf;  // C layout: vgpr r -> M=r+8*half
                const int col = n0 + jn * 16 + lr;
                float val = a[r] * alpha;
                if (bias) val += bias[col];
                const long long idx = (long long)row * ldc + col;
                if (res)  val += res[idx];
                if (Cb)   Cb[idx]   = val;
                if (Cbfb) Cbfb[idx] = (__bf16)val;
            }
        }
    }
}

// --------------------------- weight transpose-convert ----------------------
// wt[n*K + k] = bf16(w[k*N + n]), 32x32 tiles through LDS (bank-padded).
__global__ __launch_bounds__(256) void transpose_cvt_kernel(
    const float* __restrict__ w, __bf16* __restrict__ wt, int K, int N)
{
    __shared__ __bf16 tile[32][33];
    const int n0 = blockIdx.x * 32, k0 = blockIdx.y * 32;
    const int tx = threadIdx.x & 31, ty = threadIdx.x >> 5;   // 8 rows/pass
    for (int r = ty; r < 32; r += 8)
        tile[r][tx] = (__bf16)w[(long long)(k0 + r) * N + (n0 + tx)];
    __syncthreads();
    for (int r = ty; r < 32; r += 8)
        wt[(long long)(n0 + r) * K + (k0 + tx)] = tile[tx][r];
}

// ---------------------------------------------------------------------------

__global__ __launch_bounds__(256) void layernorm_kernel(
    const float* __restrict__ x, const float* __restrict__ sc,
    const float* __restrict__ bi, float* __restrict__ y,
    __bf16* __restrict__ ybf, int D)
{
    __shared__ float red[256];
    const int row = blockIdx.x;
    const float* xr = x + (long long)row * D;

    float s = 0.f;
    for (int i = threadIdx.x; i < D; i += 256) s += xr[i];
    red[threadIdx.x] = s; __syncthreads();
    for (int st = 128; st > 0; st >>= 1) {
        if (threadIdx.x < st) red[threadIdx.x] += red[threadIdx.x + st];
        __syncthreads();
    }
    const float mean = red[0] / D;
    __syncthreads();

    float v = 0.f;
    for (int i = threadIdx.x; i < D; i += 256) { float d = xr[i] - mean; v += d * d; }
    red[threadIdx.x] = v; __syncthreads();
    for (int st = 128; st > 0; st >>= 1) {
        if (threadIdx.x < st) red[threadIdx.x] += red[threadIdx.x + st];
        __syncthreads();
    }
    const float rstd = rsqrtf(red[0] / D + 1e-6f);

    float*  yr = y   + (long long)row * D;
    __bf16* yb = ybf + (long long)row * D;
    for (int i = threadIdx.x; i < D; i += 256) {
        const float o = (xr[i] - mean) * rstd * sc[i] + bi[i];
        yr[i] = o;
        yb[i] = (__bf16)o;
    }
}

// Reference multiplies scores by the causal mask (masked entries become 0,
// NOT -inf) and softmaxes the full row.  In-place fp32 + bf16 mirror.
__global__ __launch_bounds__(256) void softmax_causal_kernel(
    float* __restrict__ attn, __bf16* __restrict__ attn_bf, int S)
{
    __shared__ float red[256];
    const long long rowid = blockIdx.x;
    const int q = (int)(rowid % S);
    float*  r  = attn    + rowid * (long long)S;
    __bf16* rb = attn_bf + rowid * (long long)S;

    float m = -3.402823466e38f;
    for (int i = threadIdx.x; i < S; i += 256) {
        const float v = (i <= q) ? r[i] : 0.0f;
        r[i] = v;
        m = fmaxf(m, v);
    }
    red[threadIdx.x] = m; __syncthreads();
    for (int st = 128; st > 0; st >>= 1) {
        if (threadIdx.x < st) red[threadIdx.x] = fmaxf(red[threadIdx.x], red[threadIdx.x + st]);
        __syncthreads();
    }
    m = red[0]; __syncthreads();

    float sum = 0.f;
    for (int i = threadIdx.x; i < S; i += 256) {
        const float e = __expf(r[i] - m);
        r[i] = e;
        sum += e;
    }
    red[threadIdx.x] = sum; __syncthreads();
    for (int st = 128; st > 0; st >>= 1) {
        if (threadIdx.x < st) red[threadIdx.x] += red[threadIdx.x + st];
        __syncthreads();
    }
    const float inv = 1.0f / red[0];
    for (int i = threadIdx.x; i < S; i += 256) {
        const float v = r[i] * inv;
        r[i]  = v;
        rb[i] = (__bf16)v;
    }
}

// p [BS,3D] fp32 -> q,k bf16 (flat [B,S,D] == [B,H,S,hd] reinterpret) and
// vT bf16 laid out [B,H,hd,S] so attn@V reads B^T contiguously.
__global__ void split_qkv_kernel(const float* __restrict__ p,
                                 __bf16* __restrict__ q, __bf16* __restrict__ k,
                                 __bf16* __restrict__ vt,
                                 long long n, int D, int S, int hd)
{
    const long long SDb = (long long)S * D;
    const long long Shd = (long long)S * hd;
    const long long stride = (long long)gridDim.x * blockDim.x;
    for (long long i = (long long)blockIdx.x * blockDim.x + threadIdx.x; i < n; i += stride) {
        const int d = (int)(i % D);
        const long long rp = (i / D) * (3LL * D);
        q[i] = (__bf16)p[rp + d];
        k[i] = (__bf16)p[rp + D + d];
        const long long b = i / SDb;
        const long long f = i - b * SDb;       // = h*S*hd + s'*hd + d'
        const int h  = (int)(f / Shd);
        const long long r2 = f - (long long)h * Shd;
        const int s2 = (int)(r2 / hd);
        const int dh = (int)(r2 - (long long)s2 * hd);
        vt[((b * (D / hd) + h) * (long long)hd + dh) * S + s2] = (__bf16)p[rp + 2 * D + d];
    }
}

__global__ void add_kernel(const float* __restrict__ a, const float* __restrict__ b,
                           float* __restrict__ y, long long n)
{
    const long long stride = (long long)gridDim.x * blockDim.x;
    for (long long i = (long long)blockIdx.x * blockDim.x + threadIdx.x; i < n; i += stride)
        y[i] = a[i] + b[i];
}

__global__ void geglu_kernel(const float* __restrict__ g, __bf16* __restrict__ act,
                             long long rows, int FFdim)
{
    const long long n = rows * (long long)FFdim;
    const long long stride = (long long)gridDim.x * blockDim.x;
    for (long long i = (long long)blockIdx.x * blockDim.x + threadIdx.x; i < n; i += stride) {
        const long long row = i / FFdim;
        const int c = (int)(i - row * FFdim);
        const float* gr = g + row * (2LL * FFdim);
        const float a    = gr[c];
        const float gate = gr[FFdim + c];
        const float t = tanhf(gate * 0.7978845608f * (1.f + 0.044715f * gate * gate));
        act[i] = (__bf16)(a * 0.5f * gate * (1.f + t));
    }
}

// ---------------------------------------------------------------------------

extern "C" void kernel_launch(void* const* d_in, const int* in_sizes, int n_in,
                              void* d_out, int out_size, void* d_ws, size_t ws_size,
                              hipStream_t stream)
{
    (void)in_sizes; (void)n_in; (void)out_size; (void)ws_size;
    const int Bc = 2, Sc = 1024, Dc = 1024, Hc = 16, FFc = 4096, hd = 64;
    const int BS = Bc * Sc;
    const long long SD  = (long long)BS * Dc;            // 2,097,152
    const long long ATT = (long long)Bc * Hc * Sc * Sc;  // 33,554,432

    const float* x      = (const float*)d_in[0];
    const float* ln1_s  = (const float*)d_in[1];
    const float* ln1_b  = (const float*)d_in[2];
    const float* w_qkv1 = (const float*)d_in[3];
    const float* b_qkv1 = (const float*)d_in[4];
    const float* w_o1   = (const float*)d_in[5];
    const float* b_o1   = (const float*)d_in[6];
    const float* ln2_s  = (const float*)d_in[7];
    const float* ln2_b  = (const float*)d_in[8];
    const float* w_qkv2 = (const float*)d_in[9];
    const float* b_qkv2 = (const float*)d_in[10];
    const float* w_o2   = (const float*)d_in[11];
    const float* b_o2   = (const float*)d_in[12];
    const float* ln3_s  = (const float*)d_in[13];
    const float* ln3_b  = (const float*)d_in[14];
    const float* w1     = (const float*)d_in[15];
    const float* b1     = (const float*)d_in[16];
    const float* wg     = (const float*)d_in[17];
    const float* bg     = (const float*)d_in[18];
    const float* w2     = (const float*)d_in[19];
    const float* b2     = (const float*)d_in[20];

    // ---- workspace arena: fp32 region then bf16 region (~290 MB) ----
    float* fws = (float*)d_ws;
    long long fo = 0;
    float* ln_out  = fws + fo; fo += SD;
    float* x_res   = fws + fo; fo += SD;
    float* att_res = fws + fo; fo += SD;
    float* pbuf    = fws + fo; fo += (long long)BS * FFc;      // qkv projection p (fp32)
    float* gbuf    = fws + fo; fo += (long long)BS * 2 * FFc;  // FFN gate input

    __bf16* bws = (__bf16*)(fws + fo);
    long long bo = 0;
    __bf16* ln_bf   = bws + bo; bo += SD;
    __bf16* qb      = bws + bo; bo += SD;
    __bf16* kb      = bws + bo; bo += SD;
    __bf16* vtb     = bws + bo; bo += SD;
    __bf16* ctx_bf  = bws + bo; bo += SD;
    __bf16* hbf     = bws + bo; bo += (long long)BS * FFc;
    __bf16* act_bf  = bws + bo; bo += (long long)BS * FFc;
    __bf16* attn_bf = bws + bo; bo += ATT;                    // ALSO reused as wgT
    __bf16* wgT     = attn_bf;                                // FF*2FF == ATT elements
    __bf16* wqkv1T  = bws + bo; bo += (long long)Dc * 3 * Dc;
    __bf16* wo1T    = bws + bo; bo += (long long)Dc * Dc;
    __bf16* wqkv2T  = bws + bo; bo += (long long)Dc * 3 * Dc;
    __bf16* wo2T    = bws + bo; bo += (long long)Dc * Dc;
    __bf16* w1T     = bws + bo; bo += (long long)Dc * FFc;
    __bf16* w2T     = bws + bo; bo += (long long)FFc * Dc;

    float* out_x = (float*)d_out;
    float* attn1 = out_x + SD;
    float* attn2 = attn1 + ATT;

    auto gemm = [&](const __bf16* A, long long sA, int lda,
                    const __bf16* Bt, long long sB, int ldb,
                    float* C, __bf16* Cbf, long long sC, int ldc,
                    const float* bias, const float* res,
                    int M, int N, int K, int batch, float alpha) {
        const long long waves = (long long)batch * (M / 32) * (N / 32);
        const int blocks = (int)((waves + 7) / 8);
        hipLaunchKernelGGL(gemm_bf16_wmma, dim3(blocks), dim3(256), 0, stream,
                           A, sA, lda, Bt, sB, ldb, C, Cbf, sC, ldc,
                           bias, res, M, N, K, batch, alpha);
    };
    auto tcvt = [&](const float* w, __bf16* wt, int K, int N) {
        hipLaunchKernelGGL(transpose_cvt_kernel, dim3(N / 32, K / 32), dim3(256),
                           0, stream, w, wt, K, N);
    };
    const float inv_sqrtD = 1.0f / 32.0f;   // 1/sqrt(D), D=1024 (reference scaling)
    const int EW = 2048;

    // one-time weight transpose-converts (wg deferred: shares attn_bf region)
    tcvt(w_qkv1, wqkv1T, Dc, 3 * Dc);
    tcvt(w_o1,   wo1T,   Dc, Dc);
    tcvt(w_qkv2, wqkv2T, Dc, 3 * Dc);
    tcvt(w_o2,   wo2T,   Dc, Dc);
    tcvt(w1,     w1T,    Dc, FFc);
    tcvt(w2,     w2T,    FFc, Dc);

    // ---------------- layer 1 attention ----------------
    hipLaunchKernelGGL(layernorm_kernel, dim3(BS), dim3(256), 0, stream,
                       x, ln1_s, ln1_b, ln_out, ln_bf, Dc);
    gemm(ln_bf, 0, Dc, wqkv1T, 0, Dc, pbuf, nullptr, 0, 3 * Dc,
         b_qkv1, nullptr, BS, 3 * Dc, Dc, 1, 1.0f);
    hipLaunchKernelGGL(split_qkv_kernel, dim3(EW), dim3(256), 0, stream,
                       pbuf, qb, kb, vtb, SD, Dc, Sc, hd);
    // scores = Q @ K^T / 32 : K row-major [S,hd] is already B^T
    gemm(qb, (long long)Sc * hd, hd, kb, (long long)Sc * hd, hd,
         attn1, nullptr, (long long)Sc * Sc, Sc, nullptr, nullptr,
         Sc, Sc, hd, Bc * Hc, inv_sqrtD);
    hipLaunchKernelGGL(softmax_causal_kernel, dim3(Bc * Hc * Sc), dim3(256), 0, stream,
                       attn1, attn_bf, Sc);
    // ctx = attn @ V  (B^T = vT [hd,S]); bf16-only output feeds the projection
    gemm(attn_bf, (long long)Sc * Sc, Sc, vtb, (long long)hd * Sc, Sc,
         nullptr, ctx_bf, (long long)Sc * hd, hd, nullptr, nullptr,
         Sc, hd, Sc, Bc * Hc, 1.0f);
    // x = ln1_out + ctx @ Wo1 + b_o1 (residual fused)
    gemm(ctx_bf, 0, Dc, wo1T, 0, Dc, x_res, nullptr, 0, Dc,
         b_o1, ln_out, BS, Dc, Dc, 1, 1.0f);

    // ---------------- layer 2 attention ----------------
    hipLaunchKernelGGL(layernorm_kernel, dim3(BS), dim3(256), 0, stream,
                       x_res, ln2_s, ln2_b, ln_out, ln_bf, Dc);
    gemm(ln_bf, 0, Dc, wqkv2T, 0, Dc, pbuf, nullptr, 0, 3 * Dc,
         b_qkv2, nullptr, BS, 3 * Dc, Dc, 1, 1.0f);
    hipLaunchKernelGGL(split_qkv_kernel, dim3(EW), dim3(256), 0, stream,
                       pbuf, qb, kb, vtb, SD, Dc, Sc, hd);
    gemm(qb, (long long)Sc * hd, hd, kb, (long long)Sc * hd, hd,
         attn2, nullptr, (long long)Sc * Sc, Sc, nullptr, nullptr,
         Sc, Sc, hd, Bc * Hc, inv_sqrtD);
    hipLaunchKernelGGL(softmax_causal_kernel, dim3(Bc * Hc * Sc), dim3(256), 0, stream,
                       attn2, attn_bf, Sc);
    gemm(attn_bf, (long long)Sc * Sc, Sc, vtb, (long long)hd * Sc, Sc,
         nullptr, ctx_bf, (long long)Sc * hd, hd, nullptr, nullptr,
         Sc, hd, Sc, Bc * Hc, 1.0f);
    // att2 kept separately (final residual)
    gemm(ctx_bf, 0, Dc, wo2T, 0, Dc, att_res, nullptr, 0, Dc,
         b_o2, nullptr, BS, Dc, Dc, 1, 1.0f);
    hipLaunchKernelGGL(add_kernel, dim3(EW), dim3(256), 0, stream,
                       ln_out, att_res, x_res, SD);

    // ---------------- GEGLU FFN ----------------
    hipLaunchKernelGGL(layernorm_kernel, dim3(BS), dim3(256), 0, stream,
                       x_res, ln3_s, ln3_b, ln_out, ln_bf, Dc);
    // attn_bf is dead now -> reuse its storage for wgT
    tcvt(wg, wgT, FFc, 2 * FFc);
    // h = ln @ W1 + b1 : bf16-only output feeds FFN2
    gemm(ln_bf, 0, Dc, w1T, 0, Dc, nullptr, hbf, 0, FFc,
         b1, nullptr, BS, FFc, Dc, 1, 1.0f);
    gemm(hbf, 0, FFc, wgT, 0, FFc, gbuf, nullptr, 0, 2 * FFc,
         bg, nullptr, BS, 2 * FFc, FFc, 1, 1.0f);
    hipLaunchKernelGGL(geglu_kernel, dim3(EW), dim3(256), 0, stream,
                       gbuf, act_bf, (long long)BS, FFc);
    // final: out = act @ W2 + b2 + att2
    gemm(act_bf, 0, FFc, w2T, 0, FFc, out_x, nullptr, 0, Dc,
         b2, att_res, BS, Dc, FFc, 1, 1.0f);
}
